// I_Softmax_14370960572484
// MI455X (gfx1250) — compile-verified, matrix operand order checked by reading, not run on hardware
//
#include <hip/hip_runtime.h>
#include <hip/hip_bf16.h>

// Problem constants (match reference)
#define BROWS 2097152
#define CCOLS 16
#define NWAVES 1024            // histogram / scatter waves
#define ROWS_PER_WAVE 2048     // BROWS / NWAVES
#define ITERS 64               // ROWS_PER_WAVE / 32 lanes
#define NBLK 256               // blocks for pass1/pass3
#define NTHR 128               // threads per block (4 waves)

typedef __attribute__((ext_vector_type(16))) _Float16 v16h;
typedef __attribute__((ext_vector_type(8)))  float    v8f;

__device__ __forceinline__ float margin_xform(float x) {
  const float Mm = 3.0f + 1e-5f;
  return (x > 0.0f) ? (x * (1.0f / Mm) - 1.0f) : (x * Mm - 1.0f);
}

// ---------------- Pass 1: per-wave class histograms  hist[c*NWAVES + w] ----
__global__ void hist_kernel(const int* __restrict__ label,
                            unsigned* __restrict__ hist) {
  int gt   = blockIdx.x * blockDim.x + threadIdx.x;
  int w    = gt >> 5;
  int lane = gt & 31;
  long base = (long)w * ROWS_PER_WAVE;

  unsigned cnt[16];
#pragma unroll
  for (int c = 0; c < 16; ++c) cnt[c] = 0u;

  for (int j = 0; j < ITERS; ++j) {
    int lab = label[base + j * 32 + lane];   // coalesced b32
#pragma unroll
    for (int c = 0; c < 16; ++c) {
      unsigned bal = __builtin_amdgcn_ballot_w32(lab == c);
      cnt[c] += (unsigned)__popc(bal);       // identical across lanes
    }
  }
  if (lane == 0) {
#pragma unroll
    for (int c = 0; c < 16; ++c) hist[c * NWAVES + w] = cnt[c];
  }
}

// ---------------- Pass 2: per-class exclusive scan + class bases -----------
__global__ void scan_kernel(const unsigned* __restrict__ hist,
                            unsigned* __restrict__ offs) {
  __shared__ unsigned s[NWAVES];
  __shared__ unsigned tot[16];
  __shared__ unsigned basev[16];
  int tid = threadIdx.x;

  unsigned exc[16];
  for (int c = 0; c < 16; ++c) {
    unsigned v = hist[c * NWAVES + tid];
    unsigned acc = v;
    s[tid] = acc;
    __syncthreads();
    for (int o = 1; o < NWAVES; o <<= 1) {      // Hillis–Steele inclusive scan
      unsigned t = (tid >= o) ? s[tid - o] : 0u;
      __syncthreads();
      acc += t;
      s[tid] = acc;
      __syncthreads();
    }
    exc[c] = acc - v;                           // exclusive prefix over waves
    if (tid == NWAVES - 1) tot[c] = acc;        // class total
    __syncthreads();
  }
  if (tid == 0) {
    unsigned run = 0;
    for (int c = 0; c < 16; ++c) { basev[c] = run; run += tot[c]; }
  }
  __syncthreads();
  for (int c = 0; c < 16; ++c)
    offs[c * NWAVES + tid] = basev[c] + exc[c];
}

// -------- Pass 3: fused transform + stable scatter + loss partials ---------
__global__ void scatter_loss_kernel(const float* __restrict__ src,
                                    const int* __restrict__ label,
                                    const unsigned* __restrict__ offs,
                                    float* __restrict__ out_data,
                                    float* __restrict__ out_label,
                                    float* __restrict__ partials) {
  int gt   = blockIdx.x * blockDim.x + threadIdx.x;
  int w    = gt >> 5;
  int lane = gt & 31;
  unsigned ltmask = (1u << lane) - 1u;
  long base = (long)w * ROWS_PER_WAVE;

  unsigned cnt[16];
#pragma unroll
  for (int c = 0; c < 16; ++c) cnt[c] = offs[c * NWAVES + w];

  float lsum = 0.0f;

  for (int j = 0; j < ITERS; ++j) {
    long row = base + j * 32 + lane;
    int lab  = label[row];

    // prefetch 4 iterations (128 rows) ahead -> global_prefetch_b8
    if (j + 4 < ITERS)
      __builtin_prefetch(src + (row + 128) * CCOLS, 0, 0);

    // stable destination: classBase + wave prefix + intra-wave rank
    unsigned dst = 0;
#pragma unroll
    for (int c = 0; c < 16; ++c) {
      unsigned bal = __builtin_amdgcn_ballot_w32(lab == c);
      if (lab == c) dst = cnt[c] + (unsigned)__popc(bal & ltmask);
      cnt[c] += (unsigned)__popc(bal);
    }

    const float4* sp = (const float4*)(src + row * CCOLS);
    float4 r0 = sp[0], r1 = sp[1], r2 = sp[2], r3 = sp[3];
    float v[16] = { r0.x, r0.y, r0.z, r0.w,  r1.x, r1.y, r1.z, r1.w,
                    r2.x, r2.y, r2.z, r2.w,  r3.x, r3.y, r3.z, r3.w };

    float xy = 0.0f;
#pragma unroll
    for (int k = 0; k < 16; ++k) {       // transform only the label column
      if (k == lab) { v[k] = margin_xform(v[k]); xy = v[k]; }
    }

    // loss pieces: nll = lse - 0.9*x_y - (0.1/16)*sum(x)
    float m = v[0];
#pragma unroll
    for (int k = 1; k < 16; ++k) m = fmaxf(m, v[k]);
    float es = 0.0f, sx = 0.0f;
#pragma unroll
    for (int k = 0; k < 16; ++k) { es += __expf(v[k] - m); sx += v[k]; }
    float lse = m + __logf(es);
    lsum += lse - 0.9f * xy - 0.00625f * sx;

    float4* dp = (float4*)(out_data + (long)dst * CCOLS);
    dp[0] = make_float4(v[0],  v[1],  v[2],  v[3]);
    dp[1] = make_float4(v[4],  v[5],  v[6],  v[7]);
    dp[2] = make_float4(v[8],  v[9],  v[10], v[11]);
    dp[3] = make_float4(v[12], v[13], v[14], v[15]);
    out_label[dst] = (float)lab;
  }

  // wave butterfly reduce, then block reduce -> one partial per block
#pragma unroll
  for (int o = 16; o >= 1; o >>= 1) lsum += __shfl_xor(lsum, o, 32);
  __shared__ float wsum[NTHR / 32];
  if (lane == 0) wsum[threadIdx.x >> 5] = lsum;
  __syncthreads();
  if (threadIdx.x == 0) {
    float t = 0.0f;
    for (int q = 0; q < NTHR / 32; ++q) t += wsum[q];
    partials[blockIdx.x] = t;
  }
}

// -------- Final: 256 partials -> loss via one v_wmma_f32_16x16x32_f16 ------
// Dekker hi/lo f16 split (after exact 2^-8 prescale) x all-ones B matrix:
// f32 accumulator D holds row sums; one shfl finishes the reduction.
__global__ void final_loss_kernel(const float* __restrict__ partials,
                                  float* __restrict__ out_loss) {
  int lane = threadIdx.x;            // 32 threads, EXEC all ones (WMMA req.)
  float p[8];
#pragma unroll
  for (int j = 0; j < 8; ++j)
    p[j] = partials[lane * 8 + j] * 0.00390625f;   // exact *2^-8

  v16h a;
#pragma unroll
  for (int j = 0; j < 8; ++j) {
    _Float16 h = (_Float16)p[j];
    _Float16 l = (_Float16)(p[j] - (float)h);
    a[2 * j]     = h;
    a[2 * j + 1] = l;
  }
  v16h b;
#pragma unroll
  for (int j = 0; j < 16; ++j) b[j] = (_Float16)1.0f;
  v8f c = {};
  v8f d = __builtin_amdgcn_wmma_f32_16x16x32_f16(
      /*neg_a=*/false, a, /*neg_b=*/false, b,
      /*c_mod=*/(short)0, c, /*reuse_a=*/false, /*reuse_b=*/false);

  // per-lane: sum of 8 row-sums (rows 0-7 on lanes 0-15, rows 8-15 on 16-31)
  float s = d[0] + d[1] + d[2] + d[3] + d[4] + d[5] + d[6] + d[7];
  s += __shfl_xor(s, 16, 32);                       // grand total (scaled)
  if (lane == 0)
    *out_loss = s * (256.0f / (float)BROWS);        // undo 2^-8, mean over B
}

extern "C" void kernel_launch(void* const* d_in, const int* in_sizes, int n_in,
                              void* d_out, int out_size, void* d_ws, size_t ws_size,
                              hipStream_t stream) {
  const float* src = (const float*)d_in[0];
  const int*   lab = (const int*)d_in[1];

  float* out       = (float*)d_out;
  float* out_data  = out;                                   // B*C floats
  float* out_label = out + (long)BROWS * CCOLS;             // B floats
  float* out_loss  = out + (long)BROWS * CCOLS + BROWS;     // 1 float

  unsigned* hist     = (unsigned*)d_ws;                     // 16*1024 u32
  unsigned* offs     = hist + 16 * NWAVES;                  // 16*1024 u32
  float*    partials = (float*)(offs + 16 * NWAVES);        // 256 f32

  hist_kernel<<<NBLK, NTHR, 0, stream>>>(lab, hist);
  scan_kernel<<<1, NWAVES, 0, stream>>>(hist, offs);
  scatter_loss_kernel<<<NBLK, NTHR, 0, stream>>>(src, lab, offs,
                                                 out_data, out_label, partials);
  final_loss_kernel<<<1, 32, 0, stream>>>(partials, out_loss);
}
